// Experts_57080115364381
// MI455X (gfx1250) — compile-verified, matrix operand order checked by reading, not run on hardware
//
#include <hip/hip_runtime.h>
#include <hip/hip_bf16.h>
#include <math.h>

typedef __attribute__((ext_vector_type(16))) _Float16 v16h;
typedef __attribute__((ext_vector_type(8)))  float    v8f;

#define NEG_INF (-__builtin_inff())

constexpr int Bc = 8, Sc = 512, Dc = 768, Hc = 12, DHc = 64, Ec = 8, DFFc = 3072;
constexpr int QKVC = 3 * DHc * Hc;                 // 2304
constexpr float SCALEF = 0.03608439182435161f;     // 768^-0.5 (reference scales by D, not DH)

// ---------------------------------------------------------------------------
// WMMA helpers (gfx1250, wave32)
// ---------------------------------------------------------------------------
__device__ __forceinline__ v8f wmma(v16h a, v16h b, v8f c) {
  // v_wmma_f32_16x16x32_f16
  return __builtin_amdgcn_wmma_f32_16x16x32_f16(false, a, false, b, (short)0, c,
                                                false, false);
}

// Load a 16x32 f16 fragment (A operand, or B operand from an (N,K) row-major
// matrix) using the gfx1250 wave32 layout:
//   lanes 0-15 : row = lane,    elems 0-7 = K 0..7,  elems 8-15 = K 16..23
//   lanes 16-31: row = lane-16, elems 0-7 = K 8..15, elems 8-15 = K 24..31
// `ld` must be even (dword-aligned rows).
__device__ __forceinline__ v16h load_frag(const _Float16* p, int ld) {
  const int lane = threadIdx.x & 31;
  const _Float16* r = p + (lane & 15) * ld + ((lane < 16) ? 0 : 8);
  union { v16h h; uint32_t u[8]; } f;
#pragma unroll
  for (int i = 0; i < 4; ++i) {
    f.u[i]     = *(const uint32_t*)(r + 2 * i);
    f.u[i + 4] = *(const uint32_t*)(r + 16 + 2 * i);
  }
  return f.h;
}

// ---------------------------------------------------------------------------
// 16-lane butterfly reductions on the VALU via DPP16 (no LDS traffic).
// Rows of 16 lanes are closed under these permutes, which matches the WMMA
// C-tile layout where every matrix row lives inside one 16-lane half.
//   0xB1 = quad_perm([1,0,3,2])  (xor 1)
//   0x4E = quad_perm([2,3,0,1])  (xor 2)
//   0x141 = row_half_mirror      (combine quads within 8)
//   0x140 = row_mirror           (combine halves within 16)
// ---------------------------------------------------------------------------
template <int CTRL>
__device__ __forceinline__ float dpp_max_step(float v) {
  const int s = __builtin_amdgcn_update_dpp(0, __float_as_int(v), CTRL, 0xF, 0xF, true);
  return fmaxf(v, __int_as_float(s));
}
template <int CTRL>
__device__ __forceinline__ float dpp_add_step(float v) {
  const int s = __builtin_amdgcn_update_dpp(0, __float_as_int(v), CTRL, 0xF, 0xF, true);
  return v + __int_as_float(s);
}
__device__ __forceinline__ float red_max16(float v) {
  v = dpp_max_step<0xB1>(v);
  v = dpp_max_step<0x4E>(v);
  v = dpp_max_step<0x141>(v);
  v = dpp_max_step<0x140>(v);
  return v;
}
__device__ __forceinline__ float red_add16(float v) {
  v = dpp_add_step<0xB1>(v);
  v = dpp_add_step<0x4E>(v);
  v = dpp_add_step<0x141>(v);
  v = dpp_add_step<0x140>(v);
  return v;
}

__device__ __forceinline__ float gelu_exact(float v) {
  return 0.5f * v * (1.0f + erff(v * 0.70710678118654752f));
}

__device__ __forceinline__ float block_reduce_sum(float v) {
  __shared__ float sh[8];
  __shared__ float tot;
  __syncthreads();
#pragma unroll
  for (int d = 16; d >= 1; d >>= 1) v += __shfl_xor(v, d, 32);
  if ((threadIdx.x & 31) == 0) sh[threadIdx.x >> 5] = v;
  __syncthreads();
  if (threadIdx.x == 0) {
    float t = 0.f;
    for (int i = 0; i < (int)(blockDim.x >> 5); ++i) t += sh[i];
    tot = t;
  }
  __syncthreads();
  return tot;
}

// ---------------------------------------------------------------------------
// Elementwise utility kernels
// ---------------------------------------------------------------------------
__global__ void cvt_kernel(const float* __restrict__ in, _Float16* __restrict__ out,
                           long n) {
  long i = (long)blockIdx.x * blockDim.x + threadIdx.x;
  const long stride = (long)gridDim.x * blockDim.x;
  for (; i < n; i += stride) out[i] = (_Float16)in[i];
}

__global__ void zero_kernel(float* __restrict__ p, long n) {
  long i = (long)blockIdx.x * blockDim.x + threadIdx.x;
  const long stride = (long)gridDim.x * blockDim.x;
  for (; i < n; i += stride) p[i] = 0.f;
}

// ---------------------------------------------------------------------------
// Batched tiled GEMM:  C(M,N) = A(M,K) * W(N,K)^T  [+bias] [gelu]
// 128x128 block tile, 8 waves, each owning a 32x64 quadrant (2x4 WMMA =
// 8 v_wmma per K-step vs 6 fragment loads).  Register-staged double buffer:
// global loads for tile k+1 overlap WMMA on tile k; one barrier per K-step.
// FEAT: atomically accumulates column means into feats[e][b][:] (routing).
// SEL : expert index comes from sel[b] (post-routing FFN path).
// ---------------------------------------------------------------------------
template <int BIAS, int ACT, int OUTF16, int FEAT, int SEL>
__global__ __launch_bounds__(256) void gemm_kernel(
    const _Float16* __restrict__ A, long sAe, long sAb,
    const _Float16* __restrict__ W, long sWe,
    const float* __restrict__ bias, long sBiasE,
    float* __restrict__ C32, _Float16* __restrict__ C16,
    long sCe, long sCb, int ldc,
    float* __restrict__ feat, float featScale,
    const int* __restrict__ sel, int K) {
  int e, b;
  if (SEL) { b = blockIdx.z; e = sel[b]; }
  else     { e = blockIdx.z / Bc; b = blockIdx.z % Bc; }
  const _Float16* Ab = A + (long)e * sAe + (long)b * sAb;
  const _Float16* Wb = W + (long)e * sWe;
  const int m0 = blockIdx.x * 128, n0 = blockIdx.y * 128;
  const int wave = threadIdx.x >> 5, lane = threadIdx.x & 31;
  const int wm = (wave & 3) * 32;      // 0,32,64,96
  const int wn = (wave >> 2) * 64;     // 0,64

  __shared__ _Float16 As[2][128 * 40];  // 40-half row stride: 80B, 16B aligned
  __shared__ _Float16 Ws[2][128 * 40];

  const v8f zf = {0.f, 0.f, 0.f, 0.f, 0.f, 0.f, 0.f, 0.f};
  v8f acc[2][4];
#pragma unroll
  for (int tm = 0; tm < 2; ++tm)
#pragma unroll
    for (int tn = 0; tn < 4; ++tn) acc[tm][tn] = zf;

  const int lr = threadIdx.x >> 1;          // 0..127: tile row
  const int lc = (threadIdx.x & 1) * 16;    // 0 or 16: 16 halfs each
  const _Float16* aptr = Ab + (long)(m0 + lr) * K + lc;
  const _Float16* wptr = Wb + (long)(n0 + lr) * K + lc;

  // prologue: fetch K-slab 0 into registers
  uint4 ra0 = *(const uint4*)aptr, ra1 = *(const uint4*)(aptr + 8);
  uint4 rw0 = *(const uint4*)wptr, rw1 = *(const uint4*)(wptr + 8);

  int cur = 0;
  for (int k0 = 0; k0 < K; k0 += 32) {
    *(uint4*)(&As[cur][lr * 40 + lc])     = ra0;
    *(uint4*)(&As[cur][lr * 40 + lc + 8]) = ra1;
    *(uint4*)(&Ws[cur][lr * 40 + lc])     = rw0;
    *(uint4*)(&Ws[cur][lr * 40 + lc + 8]) = rw1;
    __syncthreads();
    if (k0 + 32 < K) {                      // overlap next slab with WMMA
      ra0 = *(const uint4*)(aptr + k0 + 32);
      ra1 = *(const uint4*)(aptr + k0 + 40);
      rw0 = *(const uint4*)(wptr + k0 + 32);
      rw1 = *(const uint4*)(wptr + k0 + 40);
      if (k0 + 64 < K) {
        __builtin_prefetch(aptr + k0 + 64, 0, 1);   // global_prefetch_b8
        __builtin_prefetch(wptr + k0 + 64, 0, 1);
      }
    }
    const v16h a0 = load_frag(&As[cur][(wm + 0)  * 40], 40);
    const v16h a1 = load_frag(&As[cur][(wm + 16) * 40], 40);
    const v16h b0 = load_frag(&Ws[cur][(wn + 0)  * 40], 40);
    const v16h b1 = load_frag(&Ws[cur][(wn + 16) * 40], 40);
    const v16h b2 = load_frag(&Ws[cur][(wn + 32) * 40], 40);
    const v16h b3 = load_frag(&Ws[cur][(wn + 48) * 40], 40);
    acc[0][0] = wmma(a0, b0, acc[0][0]);
    acc[0][1] = wmma(a0, b1, acc[0][1]);
    acc[0][2] = wmma(a0, b2, acc[0][2]);
    acc[0][3] = wmma(a0, b3, acc[0][3]);
    acc[1][0] = wmma(a1, b0, acc[1][0]);
    acc[1][1] = wmma(a1, b1, acc[1][1]);
    acc[1][2] = wmma(a1, b2, acc[1][2]);
    acc[1][3] = wmma(a1, b3, acc[1][3]);
    cur ^= 1;
  }

  // C layout: lanes 0-15 -> M = vgpr, lanes 16-31 -> M = 8+vgpr; N = lane%16.
  const int nl = lane & 15, rh = (lane < 16) ? 0 : 8;
  const long cbase = (long)e * sCe + (long)b * sCb;
#pragma unroll
  for (int tn = 0; tn < 4; ++tn) {
    const int col = n0 + wn + tn * 16 + nl;
    const float bv = BIAS ? bias[(long)e * sBiasE + col] : 0.f;
    float fsum = 0.f;
#pragma unroll
    for (int tm = 0; tm < 2; ++tm) {
#pragma unroll
      for (int j = 0; j < 8; ++j) {
        const int row = m0 + wm + tm * 16 + rh + j;
        float v = acc[tm][tn][j] + bv;
        if (ACT) v = gelu_exact(v);
        const long off = cbase + (long)row * ldc + col;
        if (OUTF16) C16[off] = (_Float16)v;
        else        C32[off] = v;
        if (FEAT) fsum += v;
      }
    }
    if (FEAT) atomicAdd(&feat[((long)e * Bc + b) * ldc + col], fsum * featScale);
  }
}

// ---------------------------------------------------------------------------
// Flash attention: one (e,b,head) x 64 query rows per workgroup; each of the
// 4 waves owns 16 query rows.  K/V tiles (32 keys) staged in LDS; V stored
// transposed so the P*V B-operand is a plain (N,K) fragment load.
// Online softmax: score C-tile rows live entirely inside one 16-lane half;
// row max/sum reduce with 4 VALU DPP16 steps (no LDS traffic).
// ---------------------------------------------------------------------------
__global__ __launch_bounds__(128) void attn_kernel(
    const _Float16* __restrict__ qkv,   // (e*B, S, 2304) : per head [q|k|v] 64 each
    const int* __restrict__ mask,       // (B, S)
    _Float16* __restrict__ ctx) {       // (e*B, S, 768)
  const int head = blockIdx.y % Hc;
  const int eb   = blockIdx.y / Hc;
  const int b    = eb % Bc;
  const _Float16* base = qkv + (long)eb * Sc * QKVC + head * (3 * DHc);
  const _Float16* Qp = base;
  const _Float16* Kp = base + DHc;
  const _Float16* Vp = base + 2 * DHc;
  const int* mrow = mask + b * Sc;
  const int wave = threadIdx.x >> 5, lane = threadIdx.x & 31;
  const int q0 = blockIdx.x * 64 + wave * 16;

  __shared__ _Float16 Ks[32 * 72];      // (key, dh)   stride 72 (144B, 16-aligned)
  __shared__ _Float16 Vt[64 * 40];      // (dh, key)   transposed V
  __shared__ _Float16 Ps[4][16 * 40];   // per-wave P scratch (C-layout -> A-layout)

  // Q fragments live in registers for the whole key loop.
  const v16h qf0 = load_frag(Qp + (long)q0 * QKVC,      QKVC);
  const v16h qf1 = load_frag(Qp + (long)q0 * QKVC + 32, QKVC);

  const v8f zf = {0.f, 0.f, 0.f, 0.f, 0.f, 0.f, 0.f, 0.f};
  v8f o[4];
#pragma unroll
  for (int t = 0; t < 4; ++t) o[t] = zf;
  float mrun[8], lrun[8];
#pragma unroll
  for (int j = 0; j < 8; ++j) { mrun[j] = NEG_INF; lrun[j] = 0.f; }

  for (int k0 = 0; k0 < Sc; k0 += 32) {
    __syncthreads();
    {
      const int r = threadIdx.x >> 2;         // 0..31 : key row
      const int c = (threadIdx.x & 3) * 16;   // 0/16/32/48 : dh chunk
      const _Float16* ks = Kp + (long)(k0 + r) * QKVC + c;
      *(uint4*)(&Ks[r * 72 + c])     = *(const uint4*)ks;
      *(uint4*)(&Ks[r * 72 + c + 8]) = *(const uint4*)(ks + 8);
      const _Float16* vs = Vp + (long)(k0 + r) * QKVC + c;
#pragma unroll
      for (int i = 0; i < 16; ++i) Vt[(c + i) * 40 + r] = vs[i];
    }
    __syncthreads();

    // scores: 16 queries x 32 keys = two 16x16 C tiles, K-dim = DH = 64
    v8f s0 = zf, s1 = zf;
    s0 = wmma(qf0, load_frag(&Ks[0],            72), s0);
    s0 = wmma(qf1, load_frag(&Ks[32],           72), s0);
    s1 = wmma(qf0, load_frag(&Ks[16 * 72],      72), s1);
    s1 = wmma(qf1, load_frag(&Ks[16 * 72 + 32], 72), s1);

    const float pen0 = (mrow[k0 + (lane & 15)]      == 0) ? NEG_INF : 0.f;
    const float pen1 = (mrow[k0 + 16 + (lane & 15)] == 0) ? NEG_INF : 0.f;
    float sa[8], sb[8], rmax[8];
#pragma unroll
    for (int j = 0; j < 8; ++j) {
      sa[j] = s0[j] * SCALEF + pen0;
      sb[j] = s1[j] * SCALEF + pen1;
      rmax[j] = red_max16(fmaxf(sa[j], sb[j]));   // DPP16 butterfly
    }

    float pa[8], pb[8], psum[8], alpha[8];
#pragma unroll
    for (int j = 0; j < 8; ++j) {
      const float mn = fmaxf(mrun[j], rmax[j]);
      const float mm = (mn == NEG_INF) ? 0.f : mn;   // fully-masked guard
      alpha[j] = __expf(mrun[j] - mm);
      pa[j] = __expf(sa[j] - mm);
      pb[j] = __expf(sb[j] - mm);
      psum[j] = red_add16(pa[j] + pb[j]);            // DPP16 butterfly
      mrun[j] = mn;
    }
#pragma unroll
    for (int j = 0; j < 8; ++j) lrun[j] = lrun[j] * alpha[j] + psum[j];
#pragma unroll
    for (int t = 0; t < 4; ++t)
#pragma unroll
      for (int j = 0; j < 8; ++j) o[t][j] = o[t][j] * alpha[j];

    // P : C-layout -> (M,K) A-layout via per-wave LDS scratch
    _Float16* Pw = Ps[wave];
    {
      const int n = lane & 15;
      const int rh = (lane < 16) ? 0 : 8;
#pragma unroll
      for (int j = 0; j < 8; ++j) {
        Pw[(rh + j) * 40 + n]      = (_Float16)pa[j];
        Pw[(rh + j) * 40 + 16 + n] = (_Float16)pb[j];
      }
    }
    asm volatile("s_wait_dscnt 0" ::: "memory");   // wave-local LDS RAW fence
    const v16h pf = load_frag(Pw, 40);
#pragma unroll
    for (int t = 0; t < 4; ++t)
      o[t] = wmma(pf, load_frag(&Vt[t * 16 * 40], 40), o[t]);
  }

  const int n = lane & 15, rh = (lane < 16) ? 0 : 8;
#pragma unroll
  for (int j = 0; j < 8; ++j) {
    const float inv = 1.f / lrun[j];
    const long row = (long)eb * Sc + q0 + rh + j;
#pragma unroll
    for (int t = 0; t < 4; ++t)
      ctx[row * Dc + head * DHc + t * 16 + n] = (_Float16)(o[t][j] * inv);
  }
}

// ---------------------------------------------------------------------------
// LayerNorm over D=768 (one row per 256-thread block, 3 elems/thread):
//   out = (A + Res - mean)/sqrt(var+eps) * g[e] + beta[e]
// ---------------------------------------------------------------------------
__global__ __launch_bounds__(256) void ln_kernel(
    const float* __restrict__ Ain, long sAe, long sAb,
    const float* __restrict__ Res, long sRb,
    const float* __restrict__ g, const float* __restrict__ bet,
    float* __restrict__ out32, _Float16* __restrict__ out16,
    const int* __restrict__ sel) {
  const int b = blockIdx.x / Sc, s = blockIdx.x % Sc;
  const int e = sel[b];
  const float* a = Ain + (long)e * sAe + (long)b * sAb + (long)s * Dc;
  const float* r = Res + (long)b * sRb + (long)s * Dc;
  float vals[3];
  float lsum = 0.f;
#pragma unroll
  for (int i = 0; i < 3; ++i) {
    const int c = threadIdx.x + i * 256;
    const float v = a[c] + r[c];
    vals[i] = v; lsum += v;
  }
  const float mean = block_reduce_sum(lsum) * (1.f / Dc);
  float vsum = 0.f;
#pragma unroll
  for (int i = 0; i < 3; ++i) { const float d = vals[i] - mean; vsum += d * d; }
  const float inv = rsqrtf(block_reduce_sum(vsum) * (1.f / Dc) + 1e-12f);
  const float* gg = g + (long)e * Dc;
  const float* bb = bet + (long)e * Dc;
#pragma unroll
  for (int i = 0; i < 3; ++i) {
    const int c = threadIdx.x + i * 256;
    const float ov = (vals[i] - mean) * inv * gg[c] + bb[c];
    const long off = ((long)b * Sc + s) * Dc + c;
    if (out32) out32[off] = ov;
    if (out16) out16[off] = (_Float16)ov;
  }
}

// ---------------------------------------------------------------------------
// Routing: squared distances (sqrt is monotonic) + first-min argmin.
// ---------------------------------------------------------------------------
__global__ __launch_bounds__(256) void dist_kernel(
    const float* __restrict__ feats, const float* __restrict__ centers,
    float* __restrict__ dist2) {
  const int e = blockIdx.x / Bc;
  const float* f = feats + (long)blockIdx.x * Dc;
  const float* c = centers + (long)e * Dc;
  float lsum = 0.f;
  for (int i = threadIdx.x; i < Dc; i += 256) {
    const float d = f[i] - c[i];
    lsum += d * d;
  }
  const float t = block_reduce_sum(lsum);
  if (threadIdx.x == 0) dist2[blockIdx.x] = t;
}

__global__ void argmin_kernel(const float* __restrict__ dist2,
                              int* __restrict__ sel) {
  const int b = threadIdx.x;
  if (b >= Bc) return;
  float best = dist2[b];
  int bi = 0;
  for (int e = 1; e < Ec; ++e) {
    const float v = dist2[e * Bc + b];
    if (v < best) { best = v; bi = e; }
  }
  sel[b] = bi;
}

// ---------------------------------------------------------------------------
// Host-side orchestration
// ---------------------------------------------------------------------------
extern "C" void kernel_launch(void* const* d_in, const int* in_sizes, int n_in,
                              void* d_out, int out_size, void* d_ws, size_t ws_size,
                              hipStream_t stream) {
  (void)in_sizes; (void)n_in; (void)out_size; (void)ws_size;
  const float* x       = (const float*)d_in[0];
  const int*   mask    = (const int*)d_in[1];
  const float* Wqkv    = (const float*)d_in[2];
  const float* Wd      = (const float*)d_in[3];
  const float* bd      = (const float*)d_in[4];
  const float* ln1_g   = (const float*)d_in[5];
  const float* ln1_b   = (const float*)d_in[6];
  const float* W1      = (const float*)d_in[7];
  const float* b1      = (const float*)d_in[8];
  const float* W2      = (const float*)d_in[9];
  const float* b2      = (const float*)d_in[10];
  const float* ln2_g   = (const float*)d_in[11];
  const float* ln2_b   = (const float*)d_in[12];
  const float* centers = (const float*)d_in[13];
  float* out = (float*)d_out;

  const long XN    = (long)Bc * Sc * Dc;
  const long WQKVN = (long)Ec * Hc * 3 * DHc * Dc;
  const long WDN   = (long)Ec * Dc * Dc;
  const long W1N   = (long)Ec * DFFc * Dc;
  const long QKVN  = (long)Ec * Bc * Sc * QKVC;
  const long CTXN  = (long)Ec * Bc * Sc * Dc;

  char* ws = (char*)d_ws;
  size_t off = 0;
  auto alloc = [&](size_t bytes) {
    size_t o = off;
    off += (bytes + 255) & ~(size_t)255;
    return o;
  };
  _Float16* Xh    = (_Float16*)(ws + alloc(XN * 2));
  _Float16* Wqkvh = (_Float16*)(ws + alloc(WQKVN * 2));
  _Float16* Wdh   = (_Float16*)(ws + alloc(WDN * 2));
  _Float16* W1h   = (_Float16*)(ws + alloc(W1N * 2));
  _Float16* W2h   = (_Float16*)(ws + alloc(W1N * 2));
  _Float16* qkvh  = (_Float16*)(ws + alloc(QKVN * 2));
  _Float16* ctxh  = (_Float16*)(ws + alloc(CTXN * 2));
  float*    att   = (float*)   (ws + alloc(CTXN * 4));
  float*    feats = (float*)   (ws + alloc((long)Ec * Bc * Dc * 4));
  float*    dist2 = (float*)   (ws + alloc((long)Ec * Bc * 4));
  int*      selp  = (int*)     (ws + alloc(Bc * 4));
  float*    h32   = (float*)   (ws + alloc(XN * 4));
  _Float16* h16   = (_Float16*)(ws + alloc(XN * 2));
  _Float16* ffnh  = (_Float16*)(ws + alloc((long)Bc * Sc * DFFc * 2));
  float*    ffn2  = (float*)   (ws + alloc(XN * 4));

  // 1) f32 -> f16 (weights then live in the 192MB L2 for all GEMM passes)
  cvt_kernel<<<2048, 256, 0, stream>>>(x, Xh, XN);
  cvt_kernel<<<2048, 256, 0, stream>>>(Wqkv, Wqkvh, WQKVN);
  cvt_kernel<<<2048, 256, 0, stream>>>(Wd, Wdh, WDN);
  cvt_kernel<<<2048, 256, 0, stream>>>(W1, W1h, W1N);
  cvt_kernel<<<2048, 256, 0, stream>>>(W2, W2h, W1N);
  zero_kernel<<<256, 256, 0, stream>>>(feats, (long)Ec * Bc * Dc);

  // 2) QKV for all (e,b): C(512,2304) = X(512,768) * Wqkv_e^T
  gemm_kernel<0, 0, 1, 0, 0><<<dim3(4, 18, 64), 256, 0, stream>>>(
      Xh, 0L, (long)Sc * Dc,
      Wqkvh, (long)Hc * 3 * DHc * Dc,
      nullptr, 0L,
      nullptr, qkvh, (long)Bc * Sc * QKVC, (long)Sc * QKVC, QKVC,
      nullptr, 0.f, nullptr, Dc);

  // 3) flash attention for all (e,b,h)
  attn_kernel<<<dim3(Sc / 64, Ec * Bc * Hc), 128, 0, stream>>>(qkvh, mask, ctxh);

  // 4) attention dense (+bias) for all (e,b); accumulates routing features
  gemm_kernel<1, 0, 0, 1, 0><<<dim3(4, 6, 64), 256, 0, stream>>>(
      ctxh, (long)Bc * Sc * Dc, (long)Sc * Dc,
      Wdh, (long)Dc * Dc,
      bd, (long)Dc,
      att, nullptr, (long)Bc * Sc * Dc, (long)Sc * Dc, Dc,
      feats, 1.f / Sc, nullptr, Dc);

  // 5) hard routing
  dist_kernel<<<Ec * Bc, 256, 0, stream>>>(feats, centers, dist2);
  argmin_kernel<<<1, 32, 0, stream>>>(dist2, selp);

  // 6) post-norm attention block for selected experts only
  ln_kernel<<<Bc * Sc, 256, 0, stream>>>(
      att, (long)Bc * Sc * Dc, (long)Sc * Dc,
      x, (long)Sc * Dc, ln1_g, ln1_b, h32, h16, selp);

  // 7) FFN (selected experts only): gelu(h W1^T + b1) W2^T + b2
  gemm_kernel<1, 1, 1, 0, 1><<<dim3(4, 24, 8), 256, 0, stream>>>(
      h16, 0L, (long)Sc * Dc,
      W1h, (long)DFFc * Dc,
      b1, (long)DFFc,
      nullptr, ffnh, 0L, (long)Sc * DFFc, DFFc,
      nullptr, 0.f, selp, Dc);
  gemm_kernel<1, 0, 0, 0, 1><<<dim3(4, 6, 8), 256, 0, stream>>>(
      ffnh, 0L, (long)Sc * DFFc,
      W2h, (long)Dc * DFFc,
      b2, (long)Dc,
      ffn2, nullptr, 0L, (long)Sc * Dc, Dc,
      nullptr, 0.f, selp, DFFc);

  // 8) post-norm FFN block -> final output
  ln_kernel<<<Bc * Sc, 256, 0, stream>>>(
      ffn2, 0L, (long)Sc * Dc,
      h32, (long)Sc * Dc, ln2_g, ln2_b, out, nullptr, selp);
}
